// CustomTransformerEncoderMoELayer_51032801411731
// MI455X (gfx1250) — compile-verified, matrix operand order checked by reading, not run on hardware
//
#include <hip/hip_runtime.h>

// ---------------------------------------------------------------------------
// Types for CDNA5 WMMA (wave32): A/B = 16 bf16 per lane, C/D = 8 f32 per lane
// ---------------------------------------------------------------------------
typedef __bf16 bf16_t;
typedef __attribute__((ext_vector_type(16))) __bf16 v16bf;
typedef __attribute__((ext_vector_type(8)))  __bf16 v8bf;
typedef __attribute__((ext_vector_type(8)))  float  v8f;
typedef unsigned int u32x4 __attribute__((ext_vector_type(4)));
typedef int          i32x4 __attribute__((ext_vector_type(4)));
typedef int          i32x8 __attribute__((ext_vector_type(8)));

// Address-space-qualified types for gfx1250 async/TDM builtins.
typedef __attribute__((address_space(3))) void lds_void;
typedef __attribute__((address_space(1))) i32x4 gbl_i32x4;
typedef __attribute__((address_space(3))) i32x4 lds_i32x4;

// Feature detection (compile-safe fallbacks)
#if defined(__has_builtin)
#  if __has_builtin(__builtin_amdgcn_tensor_load_to_lds)
#    define HAVE_TDM 1
#  endif
#  if __has_builtin(__builtin_amdgcn_global_load_async_to_lds_b128)
#    define HAVE_ASYNC 1
#  endif
#  if __has_builtin(__builtin_amdgcn_s_wait_tensorcnt)
#    define HAVE_WAIT_TENSOR 1
#  endif
#  if __has_builtin(__builtin_amdgcn_s_wait_asynccnt)
#    define HAVE_WAIT_ASYNC 1
#  endif
#endif

__device__ __forceinline__ bf16_t bf16_zero() {
    union { unsigned short s; bf16_t b; } c; c.s = 0; return c.b;
}
__device__ __forceinline__ bf16_t to_bf16(float f) {
    union { float f; unsigned int u; } a; a.f = f;
    unsigned int u = a.u + 0x7FFFu + ((a.u >> 16) & 1u);   // round-to-nearest-even
    union { unsigned short s; bf16_t b; } c; c.s = (unsigned short)(u >> 16);
    return c.b;
}
__device__ __forceinline__ float warp_sum(float v) {
    for (int o = 16; o > 0; o >>= 1) v += __shfl_xor(v, o, 32);
    return v;
}
__device__ __forceinline__ float warp_max(float v) {
    for (int o = 16; o > 0; o >>= 1) v = fmaxf(v, __shfl_xor(v, o, 32));
    return v;
}

// LDS byte offset of a __shared__ object (addrspacecast -> ptrtoint)
__device__ __forceinline__ unsigned lds_offset(void* p) {
    return (unsigned)(unsigned long long)(lds_void*)p;
}

// ---------------------------------------------------------------------------
// Tile staging primitives: TDM > async-LDS copy > plain copy
// ---------------------------------------------------------------------------
__device__ __forceinline__ void stage_b128(const bf16_t* __restrict__ g, bf16_t* l) {
#if defined(HAVE_ASYNC)
    __builtin_amdgcn_global_load_async_to_lds_b128(
        (gbl_i32x4*)g, (lds_i32x4*)l, 0, 0);
#else
    *(v8bf*)l = *(const v8bf*)g;
#endif
}

#if defined(HAVE_TDM)
// 2-D TDM tile load: tile_rows x tile_k bf16 elements, row stride in elements.
// D# layout per CDNA5 ISA (groups 0/1; groups 2/3 zero).
// Rows beyond tensor_dim1 / cols beyond tensor_dim0 are zero-filled by the TDM.
__device__ __forceinline__ void tdm_load_2d(const bf16_t* gsrc, void* ldst,
                                            int tile_k, int tile_rows,
                                            long row_stride_elems,
                                            int tensor_k, int tensor_rows)
{
    unsigned long long ga = (unsigned long long)gsrc;
    unsigned lds = lds_offset(ldst);
    u32x4 g0;
    g0.x = 1u;                                              // count=1 (valid user D#)
    g0.y = lds;                                             // lds_addr
    g0.z = (unsigned)(ga & 0xFFFFFFFFu);                    // global_addr[31:0]
    g0.w = (unsigned)((ga >> 32) & 0x01FFFFFFu) | 0x80000000u; // addr[56:32] | type=2
    union { unsigned long long q[4]; i32x8 v; } g1;
    unsigned long long d0 = (unsigned)tensor_k;             // tensor_dim0 (elems)
    unsigned long long d1 = (unsigned)tensor_rows;          // tensor_dim1
    unsigned long long s0 = (unsigned long long)row_stride_elems;
    g1.q[0] = (1ull << 16)                                  // data_size = 2 bytes
            | ((d0 & 0xFFFFull) << 48);                     // tensor_dim0[15:0]
    g1.q[1] = (d0 >> 16)                                    // tensor_dim0[31:16]
            | ((d1 & 0xFFFFFFFFull) << 16)                  // tensor_dim1
            | ((unsigned long long)(unsigned)tile_k << 48); // tile_dim0
    g1.q[2] = (unsigned long long)(unsigned)tile_rows       // tile_dim1
            | ((s0 & 0xFFFFFFFFull) << 32);                 // dim0_stride[31:0]
    g1.q[3] = (s0 >> 32) & 0xFFFFull;                       // dim0_stride[47:32]
    i32x4 z4 = {0, 0, 0, 0};
#if __clang_major__ >= 23
    i32x8 z8 = {0, 0, 0, 0, 0, 0, 0, 0};
    __builtin_amdgcn_tensor_load_to_lds(g0, g1.v, z4, z4, z8, 0);
#else
    __builtin_amdgcn_tensor_load_to_lds(g0, g1.v, z4, z4, 0);
#endif
}
#endif

// Stage a 128x32 bf16 tile whose rows are contiguous (k-major rows).
__device__ __forceinline__ void stage_tile_rows(const bf16_t* __restrict__ gbase,
                                                int ld, bf16_t (*lds)[32],
                                                int t, int wave,
                                                int remK, int remRows)
{
#if defined(HAVE_TDM)
    if (wave == 0)   // uniform branch: one TDM issue per workgroup tile
        tdm_load_2d(gbase, &lds[0][0], 32, 128, ld, remK, remRows);
#else
    const int row = t >> 2;
    const int ko  = (t & 3) << 3;
    #pragma unroll
    for (int h = 0; h < 2; h++) {
        int r = row + (h << 6);
        if (r < remRows) {
            stage_b128(gbase + (long)r * ld + ko, &lds[r][ko]);
        } else {
            v8bf z; bf16_t zz = bf16_zero();
            #pragma unroll
            for (int i = 0; i < 8; i++) z[i] = zz;
            *(v8bf*)&lds[r][ko] = z;
        }
    }
#endif
}

// Wait for outstanding tile copies (no-op for non-issuing waves), then barrier.
__device__ __forceinline__ void wait_stage()
{
#if defined(HAVE_TDM)
#  if defined(HAVE_WAIT_TENSOR)
    __builtin_amdgcn_s_wait_tensorcnt(0);
#  else
    asm volatile("s_wait_tensorcnt 0" ::: "memory");
#  endif
#elif defined(HAVE_ASYNC)
#  if defined(HAVE_WAIT_ASYNC)
    __builtin_amdgcn_s_wait_asynccnt(0);
#  else
    asm volatile("s_wait_asynccnt 0" ::: "memory");
#  endif
#endif
    __syncthreads();
}

// ---------------------------------------------------------------------------
// Fragment load from LDS. Layout per CDNA5 ISA (16-bit A 16x32):
//   lanes 0..15 hold row M=lane, K chunks [0..7] and [16..23]
//   lanes 16..31 hold row M=lane-16, K chunks [8..15] and [24..31]
// LDS tile rows are 32 bf16 (64B) so each chunk is one ds_read_b128.
// ---------------------------------------------------------------------------
__device__ __forceinline__ v16bf ld_frag(const bf16_t* base, int lane) {
    const bf16_t* p = base + (lane & 15) * 32;
    int kb = (lane >> 4) << 3;           // 0 or 8
    union { v16bf v; v8bf h[2]; } u;
    u.h[0] = *(const v8bf*)(p + kb);
    u.h[1] = *(const v8bf*)(p + kb + 16);
    return u.v;
}

// ---------------------------------------------------------------------------
// Generic bf16 GEMM:  C = epilogue(A[M,K] @ B^T + bias),  B given as [N][K].
// Tile 128x128x32, 256 threads = 8 waves; each wave: 2x4 WMMA 16x16x32 tiles.
// Double-buffered LDS; BOTH tiles staged by one TDM issue each, overlapped
// with WMMA compute. z-batching: offsets = (z/zdiv)*Hi + (z%zdiv)*Lo (elems).
// rowScale != null:  Cf[m,n] += rowScale[m] * (acc+bias)   (MoE combine)
// Assumptions: M % 128 == 0; K % 32 == 0. N may be < 128 (TDM zero-fills).
// ---------------------------------------------------------------------------
__global__ __launch_bounds__(256) void gemm_bf16_kernel(
    const bf16_t* __restrict__ A, const bf16_t* __restrict__ Bm,
    const float* __restrict__ bias,
    float* __restrict__ Cf, bf16_t* __restrict__ Cb,
    const float* __restrict__ rowScale,
    int M, int N, int K, int lda, int ldb, int ldc,
    long aHi, long aLo, long bHi, long bLo, long cHi, long cLo,
    int zdiv, int relu)
{
    __shared__ __align__(16) bf16_t As[2][128][32];
    __shared__ __align__(16) bf16_t Bs[2][128][32];   // stored [n][k]

    const int t    = threadIdx.x;
    const int lane = t & 31;
    const int wave = t >> 5;
    const int wm   = (wave >> 1) << 5;    // 0,32,64,96
    const int wn   = (wave & 1) << 6;     // 0,64

    const int z  = blockIdx.z;
    const int zh = z / zdiv, zl = z - zh * zdiv;
    A  += (long)zh * aHi + (long)zl * aLo;
    Bm += (long)zh * bHi + (long)zl * bLo;
    const long coff = (long)zh * cHi + (long)zl * cLo;

    const int m0 = blockIdx.y << 7;
    const int n0 = blockIdx.x << 7;

    v8f zf = {0.f,0.f,0.f,0.f,0.f,0.f,0.f,0.f};
    v8f acc[2][4];
    #pragma unroll
    for (int i = 0; i < 2; i++)
        #pragma unroll
        for (int j = 0; j < 4; j++) acc[i][j] = zf;

    auto stage = [&](int buf, int k0) {
        stage_tile_rows(A  + (long)m0 * lda + k0, lda, As[buf], t, wave, K - k0, M - m0);
        stage_tile_rows(Bm + (long)n0 * ldb + k0, ldb, Bs[buf], t, wave, K - k0, N - n0);
    };

    stage(0, 0);
    int cur = 0;
    for (int k0 = 0; k0 < K; k0 += 32) {
        wait_stage();                        // tile `cur` resident + barrier
        if (k0 + 32 < K) stage(cur ^ 1, k0 + 32);   // overlap next copy
        v16bf a0 = ld_frag(&As[cur][wm][0],      lane);
        v16bf a1 = ld_frag(&As[cur][wm + 16][0], lane);
        #pragma unroll
        for (int j = 0; j < 4; j++) {
            v16bf bb = ld_frag(&Bs[cur][wn + (j << 4)][0], lane);
            acc[0][j] = __builtin_amdgcn_wmma_f32_16x16x32_bf16(
                false, a0, false, bb, (short)0, acc[0][j], false, false);
            acc[1][j] = __builtin_amdgcn_wmma_f32_16x16x32_bf16(
                false, a1, false, bb, (short)0, acc[1][j], false, false);
        }
        cur ^= 1;
    }

    // ---- epilogue: C/D layout lanes 0-15 -> rows r, lanes 16-31 -> rows r+8
    const int hi8 = (lane >> 4) << 3;
    const int cl  = lane & 15;
    #pragma unroll
    for (int i = 0; i < 2; i++) {
        #pragma unroll
        for (int j = 0; j < 4; j++) {
            int col = n0 + wn + (j << 4) + cl;
            if (col >= N) continue;
            float bv = bias ? bias[col] : 0.0f;
            union { v8f v; float f[8]; } u; u.v = acc[i][j];
            #pragma unroll
            for (int r = 0; r < 8; r++) {
                int m = m0 + wm + (i << 4) + hi8 + r;
                float val = u.f[r] + bv;
                if (relu) val = fmaxf(val, 0.0f);
                long idx = coff + (long)m * ldc + col;
                if (rowScale) {
                    Cf[idx] += rowScale[m] * val;
                } else {
                    if (Cf) Cf[idx] = val;
                    if (Cb) Cb[idx] = to_bf16(val);
                }
            }
        }
    }
}

// ---------------------------------------------------------------------------
// Elementwise / transpose helpers
// ---------------------------------------------------------------------------
__global__ __launch_bounds__(256) void cvt_f32_bf16_kernel(
    const float* __restrict__ in, bf16_t* __restrict__ out, long n)
{
    long i = (long)blockIdx.x * blockDim.x + threadIdx.x;
    long stride = (long)gridDim.x * blockDim.x;
    for (; i < n; i += stride) out[i] = to_bf16(in[i]);
}

// Tiled transpose + convert: f32 in[R][C] -> bf16 out[C][R].  R,C % 32 == 0.
__global__ __launch_bounds__(256) void cvt_tr_kernel(
    const float* __restrict__ in, bf16_t* __restrict__ out, int R, int C)
{
    __shared__ float tile[32][33];
    const int cb = blockIdx.x << 5, rb = blockIdx.y << 5;
    const int tx = threadIdx.x & 31, ty = threadIdx.x >> 5;   // 32 x 8
    #pragma unroll
    for (int r = ty; r < 32; r += 8)
        tile[r][tx] = in[(long)(rb + r) * C + (cb + tx)];
    __syncthreads();
    #pragma unroll
    for (int r = ty; r < 32; r += 8)
        out[(long)(cb + r) * R + (rb + tx)] = to_bf16(tile[tx][r]);
}

// Tiled bf16 transpose per z-slab: in[z][R][C] -> out[z][C][R].
__global__ __launch_bounds__(256) void tr_bf16_kernel(
    const bf16_t* __restrict__ in, bf16_t* __restrict__ out, int R, int C)
{
    __shared__ bf16_t tile[32][33];
    const long base = (long)blockIdx.z * R * C;
    const int cb = blockIdx.x << 5, rb = blockIdx.y << 5;
    const int tx = threadIdx.x & 31, ty = threadIdx.x >> 5;
    #pragma unroll
    for (int r = ty; r < 32; r += 8)
        tile[r][tx] = in[base + (long)(rb + r) * C + (cb + tx)];
    __syncthreads();
    #pragma unroll
    for (int r = ty; r < 32; r += 8)
        out[base + (long)(cb + r) * R + (rb + tx)] = tile[tx][r];
}

__global__ __launch_bounds__(256) void zero_f32_kernel(float* __restrict__ p, long n)
{
    long i = (long)blockIdx.x * blockDim.x + threadIdx.x;
    long stride = (long)gridDim.x * blockDim.x;
    for (; i < n; i += stride) p[i] = 0.0f;
}

__global__ void sum64_kernel(const float* __restrict__ v, float* __restrict__ out)
{
    float x = v[threadIdx.x];
    x = warp_sum(x);
    __shared__ float s[2];
    if ((threadIdx.x & 31) == 0) s[threadIdx.x >> 5] = x;
    __syncthreads();
    if (threadIdx.x == 0) out[0] = s[0] + s[1];
}

// ---------------------------------------------------------------------------
// Softmax over keys with the frac-based additive term:
//   v = (logit + ((fj-fi)/(fi*fj+eps)) * sum(attn_bias)) * scale
// One block per (b,h,i) row of T=1024 keys; emits bf16 probabilities.
// ---------------------------------------------------------------------------
__global__ __launch_bounds__(256) void attn_softmax_kernel(
    const float* __restrict__ logits, bf16_t* __restrict__ probs,
    const float* __restrict__ frac, const float* __restrict__ sabp,
    int T, int H, float scale)
{
    const int rowid = blockIdx.x;                 // b*H*T + h*T + i
    const int i = rowid % T;
    const int b = rowid / (T * H);
    const float* lr = logits + (long)rowid * T;
    bf16_t* pr = probs + (long)rowid * T;
    const float* fb = frac + (long)b * T;
    const float fi = fb[i];
    const float sab = sabp[0];
    const int t = threadIdx.x, lane = t & 31, wave = t >> 5;

    float vals[4];
    float mx = -3.4e38f;
    #pragma unroll
    for (int c = 0; c < 4; c++) {
        int j = t + (c << 8);
        float fj = fb[j];
        float fac = (fj - fi) / (fi * fj + 1e-8f);
        float v = (lr[j] + fac * sab) * scale;
        vals[c] = v;
        mx = fmaxf(mx, v);
    }
    __shared__ float sred[8];
    float wmx = warp_max(mx);
    if (lane == 0) sred[wave] = wmx;
    __syncthreads();
    float bm = sred[0];
    for (int w = 1; w < 8; w++) bm = fmaxf(bm, sred[w]);
    __syncthreads();

    float sum = 0.f;
    #pragma unroll
    for (int c = 0; c < 4; c++) { float e = __expf(vals[c] - bm); vals[c] = e; sum += e; }
    float ws = warp_sum(sum);
    if (lane == 0) sred[wave] = ws;
    __syncthreads();
    float tot = 0.f;
    for (int w = 0; w < 8; w++) tot += sred[w];
    float inv = 1.0f / tot;
    #pragma unroll
    for (int c = 0; c < 4; c++) pr[t + (c << 8)] = to_bf16(vals[c] * inv);
}

// ---------------------------------------------------------------------------
// y = LayerNorm(a + b) * g + beta ; optional bf16 copy for next GEMM input
// ---------------------------------------------------------------------------
__global__ __launch_bounds__(256) void add_ln_kernel(
    const float* __restrict__ a, const float* __restrict__ b,
    const float* __restrict__ g, const float* __restrict__ be,
    float* __restrict__ yf, bf16_t* __restrict__ yb, int D)
{
    const long row = blockIdx.x;
    const float* ar = a + row * D;
    const float* br = b + row * D;
    const int t = threadIdx.x, lane = t & 31, wave = t >> 5;

    float v[4]; float s = 0.f;
    #pragma unroll
    for (int c = 0; c < 4; c++) { int j = t + (c << 8); float x = ar[j] + br[j]; v[c] = x; s += x; }

    __shared__ float sred[8];
    float wsum = warp_sum(s);
    if (lane == 0) sred[wave] = wsum;
    __syncthreads();
    float tot = 0.f;
    for (int w = 0; w < 8; w++) tot += sred[w];
    float mu = tot / (float)D;
    __syncthreads();

    float s2 = 0.f;
    #pragma unroll
    for (int c = 0; c < 4; c++) { float d = v[c] - mu; s2 += d * d; }
    float w2 = warp_sum(s2);
    if (lane == 0) sred[wave] = w2;
    __syncthreads();
    float tot2 = 0.f;
    for (int w = 0; w < 8; w++) tot2 += sred[w];
    float rr = rsqrtf(tot2 / (float)D + 1e-5f);

    #pragma unroll
    for (int c = 0; c < 4; c++) {
        int j = t + (c << 8);
        float y = (v[c] - mu) * rr * g[j] + be[j];
        yf[row * D + j] = y;
        if (yb) yb[row * D + j] = to_bf16(y);
    }
}

// ---------------------------------------------------------------------------
// Gate: s = softmax(x @ gate_w + gate_b); top-2 -> combine weights,
// stored expert-major: cwT[e*ntok + token]
// ---------------------------------------------------------------------------
__global__ __launch_bounds__(128) void gate_kernel(
    const float* __restrict__ x, const float* __restrict__ gw,
    const float* __restrict__ gb, float* __restrict__ cwT, int D, int ntok)
{
    const int tok = blockIdx.x;
    const float* xr = x + (long)tok * D;
    const int t = threadIdx.x, lane = t & 31, wave = t >> 5;

    float p[4] = {0.f, 0.f, 0.f, 0.f};
    for (int d = t; d < D; d += 128) {
        float xv = xr[d];
        const float* g = gw + d * 4;
        p[0] += xv * g[0]; p[1] += xv * g[1]; p[2] += xv * g[2]; p[3] += xv * g[3];
    }
    __shared__ float red[4][4];
    #pragma unroll
    for (int e = 0; e < 4; e++) p[e] = warp_sum(p[e]);
    if (lane == 0) { for (int e = 0; e < 4; e++) red[wave][e] = p[e]; }
    __syncthreads();
    if (t == 0) {
        float s[4];
        for (int e = 0; e < 4; e++)
            s[e] = gb[e] + red[0][e] + red[1][e] + red[2][e] + red[3][e];
        float mx = fmaxf(fmaxf(s[0], s[1]), fmaxf(s[2], s[3]));
        float ex[4], tot = 0.f;
        for (int e = 0; e < 4; e++) { ex[e] = __expf(s[e] - mx); tot += ex[e]; }
        for (int e = 0; e < 4; e++) ex[e] /= tot;
        int i0 = 0;
        for (int e = 1; e < 4; e++) if (ex[e] > ex[i0]) i0 = e;
        int i1 = -1;
        for (int e = 0; e < 4; e++) {
            if (e == i0) continue;
            if (i1 < 0 || ex[e] > ex[i1]) i1 = e;
        }
        for (int e = 0; e < 4; e++)
            cwT[(long)e * ntok + tok] = (e == i0 || e == i1) ? ex[e] : 0.0f;
    }
}

// ---------------------------------------------------------------------------
// Host orchestration
// ---------------------------------------------------------------------------
extern "C" void kernel_launch(void* const* d_in, const int* in_sizes, int n_in,
                              void* d_out, int out_size, void* d_ws, size_t ws_size,
                              hipStream_t stream)
{
    const int Bb = 4, T = 1024, D = 1024, H = 16, HD = 64, FF = 4096, E = 4;
    const int NTOK = Bb * T;

    const float* src       = (const float*)d_in[0];
    const float* frac      = (const float*)d_in[1];
    const float* Wq        = (const float*)d_in[2];
    const float* bq        = (const float*)d_in[3];
    const float* Wk        = (const float*)d_in[4];
    const float* bk        = (const float*)d_in[5];
    const float* Wv        = (const float*)d_in[6];
    const float* bv        = (const float*)d_in[7];
    const float* attn_bias = (const float*)d_in[8];
    const float* Wo        = (const float*)d_in[9];
    const float* bo        = (const float*)d_in[10];
    const float* gate_w    = (const float*)d_in[11];
    const float* gate_b    = (const float*)d_in[12];
    const float* ew1       = (const float*)d_in[13];
    const float* eb1       = (const float*)d_in[14];
    const float* ew2       = (const float*)d_in[15];
    const float* eb2       = (const float*)d_in[16];
    const float* ln1_g     = (const float*)d_in[17];
    const float* ln1_b     = (const float*)d_in[18];
    const float* ln2_g     = (const float*)d_in[19];
    const float* ln2_b     = (const float*)d_in[20];
    float* out = (float*)d_out;

    char* ws = (char*)d_ws;
    size_t off = 0;
    auto alloc = [&](size_t bytes) -> char* {
        size_t cur = (off + 255) & ~(size_t)255;
        off = cur + bytes;
        return ws + cur;
    };

    bf16_t* sSrc  = (bf16_t*)alloc((size_t)NTOK * D * 2);
    bf16_t* sWqT  = (bf16_t*)alloc((size_t)D * D * 2);      // [N][K]
    bf16_t* sWkT  = (bf16_t*)alloc((size_t)D * D * 2);
    bf16_t* sWvT  = (bf16_t*)alloc((size_t)D * D * 2);
    bf16_t* sWoT  = (bf16_t*)alloc((size_t)D * D * 2);
    bf16_t* sEw1T = (bf16_t*)alloc((size_t)E * FF * D * 2); // per e: [FF][D]
    bf16_t* sEw2T = (bf16_t*)alloc((size_t)E * D * FF * 2); // per e: [D][FF]
    bf16_t* Qb    = (bf16_t*)alloc((size_t)NTOK * D * 2);
    bf16_t* Kb    = (bf16_t*)alloc((size_t)NTOK * D * 2);
    bf16_t* Vb    = (bf16_t*)alloc((size_t)NTOK * D * 2);
    bf16_t* Vt    = (bf16_t*)alloc((size_t)NTOK * D * 2);   // per b: [D][T]
    float*  logits = (float*)alloc((size_t)Bb * H * T * T * 4);
    bf16_t* probs  = (bf16_t*)alloc((size_t)Bb * H * T * T * 2);
    bf16_t* ctxb   = (bf16_t*)alloc((size_t)NTOK * D * 2);
    float*  attnout= (float*)alloc((size_t)NTOK * D * 4);
    float*  xf     = (float*)alloc((size_t)NTOK * D * 4);
    bf16_t* xb     = (bf16_t*)alloc((size_t)NTOK * D * 2);
    float*  sab    = (float*)alloc(256);
    float*  cwT    = (float*)alloc((size_t)E * NTOK * 4);
    bf16_t* hb     = (bf16_t*)alloc((size_t)NTOK * FF * 2);
    float*  ff     = (float*)alloc((size_t)NTOK * D * 4);

    auto cvt = [&](const float* in, bf16_t* o, long n) {
        cvt_f32_bf16_kernel<<<2048, 256, 0, stream>>>(in, o, n);
    };
    auto cvt_tr = [&](const float* in, bf16_t* o, int R, int C) {
        dim3 grid(C / 32, R / 32, 1);
        cvt_tr_kernel<<<grid, 256, 0, stream>>>(in, o, R, C);
    };
    auto gemm = [&](const bf16_t* A, const bf16_t* Bmat, const float* bias,
                    float* Cf, bf16_t* Cb, const float* rs,
                    int M, int N, int K, int lda, int ldb, int ldc,
                    long aHi, long aLo, long bHi, long bLo, long cHi, long cLo,
                    int zdiv, int nz, int relu) {
        dim3 grid((N + 127) / 128, (M + 127) / 128, nz);
        gemm_bf16_kernel<<<grid, 256, 0, stream>>>(
            A, Bmat, bias, Cf, Cb, rs, M, N, K, lda, ldb, ldc,
            aHi, aLo, bHi, bLo, cHi, cLo, zdiv, relu);
    };

    // 1) f32 -> bf16 conversions; weights transposed to [N][K] for TDM staging
    cvt(src, sSrc, (long)NTOK * D);
    cvt_tr(Wq, sWqT, D, D);
    cvt_tr(Wk, sWkT, D, D);
    cvt_tr(Wv, sWvT, D, D);
    cvt_tr(Wo, sWoT, D, D);
    for (int e = 0; e < E; e++) {
        cvt_tr(ew1 + (size_t)e * D * FF, sEw1T + (size_t)e * FF * D, D, FF);
        cvt_tr(ew2 + (size_t)e * FF * D, sEw2T + (size_t)e * D * FF, FF, D);
    }

    // 2) sum(attn_bias)
    sum64_kernel<<<1, 64, 0, stream>>>(attn_bias, sab);

    // 3) QKV projections (bf16 outputs)
    gemm(sSrc, sWqT, bq, nullptr, Qb, nullptr, NTOK, D, D, D, D, D,
         0, 0, 0, 0, 0, 0, 1, 1, 0);
    gemm(sSrc, sWkT, bk, nullptr, Kb, nullptr, NTOK, D, D, D, D, D,
         0, 0, 0, 0, 0, 0, 1, 1, 0);
    gemm(sSrc, sWvT, bv, nullptr, Vb, nullptr, NTOK, D, D, D, D, D,
         0, 0, 0, 0, 0, 0, 1, 1, 0);

    // 3b) V -> V^T per batch: [T][D] -> [D][T]  (rows = h*64+d, cols = token)
    {
        dim3 grid(D / 32, T / 32, Bb);
        tr_bf16_kernel<<<grid, 256, 0, stream>>>(Vb, Vt, T, D);
    }

    // 4) logits = Q @ K^T per (b,h): z = b*H+h  (K matrix rows are k-contiguous)
    gemm(Qb, Kb, nullptr, logits, nullptr, nullptr, T, T, HD, D, D, T,
         (long)T * D, 64, (long)T * D, 64, (long)H * T * T, (long)T * T,
         H, Bb * H, 0);

    // 5) softmax with frac/attn_bias term -> bf16 probs
    attn_softmax_kernel<<<Bb * H * T, 256, 0, stream>>>(
        logits, probs, frac, sab, T, H, 0.125f /* 1/sqrt(64) */);

    // 6) attn = P @ V per (b,h): B = V^T slab [64][T]  (N=64; TDM zero-fills)
    gemm(probs, Vt, nullptr, nullptr, ctxb, nullptr, T, HD, T, T, T, D,
         (long)H * T * T, (long)T * T, (long)D * T, (long)HD * T,
         (long)T * D, 64, H, Bb * H, 0);

    // 7) output projection
    gemm(ctxb, sWoT, bo, attnout, nullptr, nullptr, NTOK, D, D, D, D, D,
         0, 0, 0, 0, 0, 0, 1, 1, 0);

    // 8) x = LN(src + attn)
    add_ln_kernel<<<NTOK, 256, 0, stream>>>(src, attnout, ln1_g, ln1_b, xf, xb, D);

    // 9) gating -> combine weights (expert-major)
    gate_kernel<<<NTOK, 128, 0, stream>>>(xf, gate_w, gate_b, cwT, D, NTOK);

    // 10) ff = sum_e combine_e * (relu(x@ew1_e+eb1_e)@ew2_e + eb2_e)
    zero_f32_kernel<<<2048, 256, 0, stream>>>(ff, (long)NTOK * D);
    for (int e = 0; e < E; e++) {
        gemm(xb, sEw1T + (size_t)e * FF * D, eb1 + (size_t)e * FF,
             nullptr, hb, nullptr, NTOK, FF, D, D, D, FF,
             0, 0, 0, 0, 0, 0, 1, 1, 1);
        gemm(hb, sEw2T + (size_t)e * D * FF, eb2 + (size_t)e * D,
             ff, nullptr, cwT + (size_t)e * NTOK, NTOK, D, FF, FF, FF, D,
             0, 0, 0, 0, 0, 0, 1, 1, 0);
    }

    // 11) out = LN(x + ff)
    add_ln_kernel<<<NTOK, 256, 0, stream>>>(xf, ff, ln2_g, ln2_b, out, nullptr, D);
}
